// VisionAttention_39204461478552
// MI455X (gfx1250) — matrix operand order repaired and checked by reading, not compiled, not run on hardware
//
#include <hip/hip_runtime.h>
#include <hip/hip_bf16.h>

// ---------------------------------------------------------------------------
// VisionAttention on MI455X (gfx1250, wave32, WMMA)
//   S=3136 seq, H=16 heads, D=80 head-dim, HID=1280
//   ~92 GFLOP vs ~60MB traffic -> compute bound -> bf16 WMMA everywhere.
//   Attention K/V tiles staged to LDS with async DMA (ASYNCcnt), double
//   buffered so the tensor pipe never waits on HBM/L2.
// ---------------------------------------------------------------------------

#define SEQ   3136
#define NH    16
#define HD    80
#define HID   1280
#define DPAD  96              // 80 padded to 3 * K32 chunks
#define KSTEP 32              // bf16 WMMA K
#define ATT_SCALE 0.1118033988749895f  // 1/sqrt(80)

typedef __bf16 bf16_t;
typedef __attribute__((ext_vector_type(16))) __bf16 v16bf;
typedef __attribute__((ext_vector_type(8)))  __bf16 v8bf;
typedef __attribute__((ext_vector_type(8)))  float  v8f;
typedef __attribute__((ext_vector_type(4)))  int    i32x4;

#define AS1 __attribute__((address_space(1)))
#define AS3 __attribute__((address_space(3)))

#if __has_builtin(__builtin_amdgcn_global_load_async_to_lds_b128)
#define HAVE_ASYNC_LDS 1
#else
#define HAVE_ASYNC_LDS 0
#endif

#if HAVE_ASYNC_LDS
#if __has_builtin(__builtin_amdgcn_s_wait_asynccnt)
#define WAIT_ASYNC() __builtin_amdgcn_s_wait_asynccnt(0)
#else
#define WAIT_ASYNC() asm volatile("s_wait_asynccnt 0" ::: "memory")
#endif
#else
#define WAIT_ASYNC() do {} while (0)
#endif

// Move 16 bytes global -> LDS. Async (DMA, ASYNCcnt) when available,
// otherwise synchronous through a VGPR (DScnt path).
__device__ __forceinline__ void stage16(const bf16_t* g, bf16_t* l) {
#if HAVE_ASYNC_LDS
    __builtin_amdgcn_global_load_async_to_lds_b128(
        (AS1 i32x4*)(void*)g, (AS3 i32x4*)l, 0, 0);
#else
    *(v8bf*)l = *(const v8bf*)g;
#endif
}

__device__ __forceinline__ bf16_t f2bf(float f) {
    unsigned u = __builtin_bit_cast(unsigned, f);
    unsigned r = u + 0x7FFFu + ((u >> 16) & 1u);   // round-to-nearest-even
    unsigned short h = (unsigned short)(r >> 16);
    return __builtin_bit_cast(bf16_t, h);
}

__device__ __forceinline__ v16bf combine16(v8bf lo, v8bf hi) {
    v16bf r;
#pragma unroll
    for (int i = 0; i < 8; ++i) { r[i] = lo[i]; r[i + 8] = hi[i]; }
    return r;
}

#define WMMA_BF16(A, B, C) \
    __builtin_amdgcn_wmma_f32_16x16x32_bf16(false, (A), false, (B), (short)0, (C), false, false)

// ---------------------------------------------------------------------------
// Kernel 1: f32 -> bf16 conversion
// ---------------------------------------------------------------------------
__global__ void cvt_f32_bf16(const float* __restrict__ src,
                             bf16_t* __restrict__ dst, int n) {
    int i = blockIdx.x * blockDim.x + threadIdx.x;
    if (i < n) dst[i] = f2bf(src[i]);
}

// ---------------------------------------------------------------------------
// Kernel 2: QKV GEMM  qkv = x @ W^T + b   (M=3136, N=3840, K=1280)
//   8 waves/WG -> 32x64 tile, 1 wave = one 16x16 C tile, 40 K-iterations.
//   Epilogue scatters q,k into [H,S,96] (raw, pre-RoPE) and v into [H,D,S].
// ---------------------------------------------------------------------------
__global__ __launch_bounds__(256) void qkv_gemm(
    const bf16_t* __restrict__ xb,    // [SEQ, HID]
    const bf16_t* __restrict__ wb,    // [3*HID, HID]
    const float*  __restrict__ bias,  // [3*HID]
    bf16_t* __restrict__ qp,          // [NH, SEQ, DPAD]
    bf16_t* __restrict__ kp,          // [NH, SEQ, DPAD]
    bf16_t* __restrict__ vt)          // [NH, HD, SEQ]
{
    const int lane = threadIdx.x & 31;
    const int w    = threadIdx.x >> 5;           // 0..7
    const int l16  = lane & 15, lhi = lane >> 4;
    const int mb   = (blockIdx.y * 2 + (w >> 2)) * 16;
    const int nb   = (blockIdx.x * 4 + (w & 3)) * 16;

    const bf16_t* arow = xb + (size_t)(mb + l16) * HID;
    const bf16_t* brow = wb + (size_t)(nb + l16) * HID;

    v8f acc = {};
#pragma unroll 4
    for (int kb = 0; kb < HID; kb += KSTEP) {
        __builtin_prefetch(brow + kb + 8 * KSTEP, 0, 0);   // global_prefetch_b8
        v8bf alo = *(const v8bf*)(arow + kb + lhi * 8);
        v8bf ahi = *(const v8bf*)(arow + kb + 16 + lhi * 8);
        v8bf blo = *(const v8bf*)(brow + kb + lhi * 8);
        v8bf bhi = *(const v8bf*)(brow + kb + 16 + lhi * 8);
        acc = WMMA_BF16(combine16(alo, ahi), combine16(blo, bhi), acc);
    }

    const int n  = nb + l16;
    const float bn = bias[n];
#pragma unroll
    for (int r = 0; r < 8; ++r) {
        const int s  = mb + r + lhi * 8;
        const float v = acc[r] + bn;
        if (n < HID) {                       // Q (raw, RoPE applied later)
            const int h = n / HD, d = n % HD;
            qp[((size_t)h * SEQ + s) * DPAD + d] = f2bf(v);
        } else if (n < 2 * HID) {            // K
            const int nn = n - HID, h = nn / HD, d = nn % HD;
            kp[((size_t)h * SEQ + s) * DPAD + d] = f2bf(v);
        } else {                             // V, transposed [H, D, S]
            const int nn = n - 2 * HID, h = nn / HD, d = nn % HD;
            vt[((size_t)h * HD + d) * SEQ + s] = f2bf(v);
        }
    }
}

// ---------------------------------------------------------------------------
// Kernel 3: in-place RoPE on q,k (+fold 1/sqrt(D) into q) and zero pad cols.
// ---------------------------------------------------------------------------
__global__ void rope_kernel(bf16_t* __restrict__ qp, bf16_t* __restrict__ kp,
                            const float* __restrict__ cosT,   // [SEQ, HD]
                            const float* __restrict__ sinT)   // [SEQ, HD]
{
    const size_t tid   = (size_t)blockIdx.x * blockDim.x + threadIdx.x;
    const size_t total = (size_t)2 * NH * SEQ * 48;
    if (tid >= total) return;
    const int    slot = (int)(tid % 48);
    const size_t rest = tid / 48;
    const int    s    = (int)(rest % SEQ);
    const size_t hh   = rest / SEQ;          // 0..2*NH-1
    const int    isK  = hh >= NH;
    const int    h    = isK ? (int)hh - NH : (int)hh;
    bf16_t* base = (isK ? kp : qp) + ((size_t)h * SEQ + s) * DPAD;

    if (slot >= 40) {                         // zero pad [80,96)
        const int pd = 80 + (slot - 40) * 2;
        base[pd] = f2bf(0.0f); base[pd + 1] = f2bf(0.0f);
        return;
    }
    const int d = slot;                       // 0..39, partner d+40
    const float a  = (float)base[d];
    const float b  = (float)base[d + 40];
    const float c0 = cosT[(size_t)s * HD + d];
    const float s0 = sinT[(size_t)s * HD + d];
    const float c1 = cosT[(size_t)s * HD + d + 40];
    const float s1 = sinT[(size_t)s * HD + d + 40];
    float o0 = a * c0 - b * s0;               // rotate_half: -x2 in low half
    float o1 = b * c1 + a * s1;               //               x1 in high half
    if (!isK) { o0 *= ATT_SCALE; o1 *= ATT_SCALE; }
    base[d]      = f2bf(o0);
    base[d + 40] = f2bf(o1);
}

// ---------------------------------------------------------------------------
// Kernel 4: flash attention. 1 wave = 16 query rows of one head; 4 waves/WG.
//   Per 32-key block: K tile (32x96) + V tile (80x32) staged to LDS with
//   async DMA, double-buffered; scores = 2x3 WMMAs from LDS fragments;
//   online softmax (shfl width 16); P transposed via LDS; 5 PV WMMAs.
// ---------------------------------------------------------------------------
__global__ __launch_bounds__(128) void attn_kernel(
    const bf16_t* __restrict__ qp,    // [NH, SEQ, DPAD]  (rope'd, scaled)
    const bf16_t* __restrict__ kp,    // [NH, SEQ, DPAD]  (rope'd)
    const bf16_t* __restrict__ vt,    // [NH, HD, SEQ]
    bf16_t* __restrict__ attn)        // [SEQ, HID]
{
    __shared__ bf16_t ksh[2][32 * DPAD];      // 2 x 6144 B
    __shared__ bf16_t vsh[2][HD * 32];        // 2 x 5120 B
    __shared__ bf16_t psh[4 * 16 * 32];       // per-wave P staging, 4096 B

    const int tid  = threadIdx.x;
    const int lane = tid & 31;
    const int w    = tid >> 5;                // 0..3
    const int l16  = lane & 15, lhi = lane >> 4;
    const int h    = blockIdx.y;
    const int mb   = (blockIdx.x * 4 + w) * 16;

    const bf16_t* khead = kp + (size_t)h * SEQ * DPAD;
    const bf16_t* vhead = vt + (size_t)h * HD * SEQ;

    // cooperative stagers: K tile is one contiguous 6144B block; V tile is
    // 80 rows of 64B at stride SEQ.
    auto stageK = [&](int jb, int buf) {
#pragma unroll
        for (int t = 0; t < 3; ++t) {
            const int c = tid + t * 128;          // 384 x 16B chunks
            stage16(khead + (size_t)jb * DPAD + c * 8, &ksh[buf][c * 8]);
        }
    };
    auto stageV = [&](int jb, int buf) {
#pragma unroll
        for (int t = 0; t < 3; ++t) {
            const int c = tid + t * 128;          // 320 x 16B chunks
            if (c < 320) {
                const int row = c >> 2, col = (c & 3) * 8;
                stage16(vhead + (size_t)row * SEQ + jb + col,
                        &vsh[buf][row * 32 + col]);
            }
        }
    };

    // Preload the 16x96 Q tile as 3 A-fragments (stays in VGPRs all loop).
    v16bf aq[3];
    const bf16_t* qrow = qp + ((size_t)h * SEQ + (mb + l16)) * DPAD;
#pragma unroll
    for (int c = 0; c < 3; ++c) {
        v8bf lo = *(const v8bf*)(qrow + c * 32 + lhi * 8);
        v8bf hi = *(const v8bf*)(qrow + c * 32 + 16 + lhi * 8);
        aq[c] = combine16(lo, hi);
    }

    float m[8], l[8];
    v8f o[5];
#pragma unroll
    for (int r = 0; r < 8; ++r) { m[r] = -1e30f; l[r] = 0.0f; }
#pragma unroll
    for (int t = 0; t < 5; ++t) o[t] = v8f{};

    bf16_t* pw = psh + w * 16 * 32;

    stageK(0, 0);                              // prologue fill, buffer 0
    stageV(0, 0);
    int cur = 0;

    for (int j = 0; j < SEQ; j += 32) {
        WAIT_ASYNC();                          // own async fills landed
        __syncthreads();                       // all waves' fills visible
        if (j + 32 < SEQ) {                    // DMA next block while computing
            stageK(j + 32, cur ^ 1);
            stageV(j + 32, cur ^ 1);
        }
        const bf16_t* kt = ksh[cur];
        const bf16_t* vtl = vsh[cur];

        // ---- scores: two 16x16 tiles over keys [j, j+32) (from LDS) ----
        v8f sc0 = {}, sc1 = {};
#pragma unroll
        for (int c = 0; c < 3; ++c) {
            const bf16_t* kr0 = kt + (size_t)(l16) * DPAD + c * 32 + lhi * 8;
            const bf16_t* kr1 = kt + (size_t)(16 + l16) * DPAD + c * 32 + lhi * 8;
            v16bf b0 = combine16(*(const v8bf*)kr0, *(const v8bf*)(kr0 + 16));
            v16bf b1 = combine16(*(const v8bf*)kr1, *(const v8bf*)(kr1 + 16));
            sc0 = WMMA_BF16(aq[c], b0, sc0);
            sc1 = WMMA_BF16(aq[c], b1, sc1);
        }

        // ---- online softmax (row stats replicated across 16 lanes) ----
#pragma unroll
        for (int r = 0; r < 8; ++r) {
            float mx = fmaxf(sc0[r], sc1[r]);
#pragma unroll
            for (int t = 1; t < 16; t <<= 1) mx = fmaxf(mx, __shfl_xor(mx, t, 16));
            const float mn  = fmaxf(m[r], mx);
            const float rsc = __expf(m[r] - mn);
            m[r] = mn;
            const float p0 = __expf(sc0[r] - mn);
            const float p1 = __expf(sc1[r] - mn);
            sc0[r] = p0; sc1[r] = p1;
            float ps = p0 + p1;
#pragma unroll
            for (int t = 1; t < 16; t <<= 1) ps += __shfl_xor(ps, t, 16);
            l[r] = l[r] * rsc + ps;
#pragma unroll
            for (int dt = 0; dt < 5; ++dt) o[dt][r] *= rsc;
        }

        // ---- transpose P: C-fragment layout -> A-fragment layout via LDS ----
#pragma unroll
        for (int r = 0; r < 8; ++r) {
            const int row = r + lhi * 8;
            pw[row * 32 + l16]      = f2bf(sc0[r]);
            pw[row * 32 + 16 + l16] = f2bf(sc1[r]);
        }
        asm volatile("s_wait_dscnt 0" ::: "memory");   // in-wave LDS RAW
        const bf16_t* prow = pw + l16 * 32;
        v16bf pa = combine16(*(const v8bf*)(prow + lhi * 8),
                             *(const v8bf*)(prow + 16 + lhi * 8));

        // ---- O += P @ V : 5 N-tiles over head dim (V from LDS) ----
#pragma unroll
        for (int dt = 0; dt < 5; ++dt) {
            const bf16_t* vr = vtl + (dt * 16 + l16) * 32 + lhi * 8;
            v16bf bv = combine16(*(const v8bf*)vr, *(const v8bf*)(vr + 16));
            o[dt] = WMMA_BF16(pa, bv, o[dt]);
        }
        cur ^= 1;
    }

    // ---- normalize and store [S, HID] bf16 ----
#pragma unroll
    for (int r = 0; r < 8; ++r) {
        const int s = mb + r + lhi * 8;
        const float inv = 1.0f / l[r];
#pragma unroll
        for (int dt = 0; dt < 5; ++dt)
            attn[(size_t)s * HID + h * HD + dt * 16 + l16] = f2bf(o[dt][r] * inv);
    }
}

// ---------------------------------------------------------------------------
// Kernel 5: projection GEMM  out = attn @ Wp^T + b + hidden_states  (f32 out)
// ---------------------------------------------------------------------------
__global__ __launch_bounds__(256) void proj_gemm(
    const bf16_t* __restrict__ ab,     // [SEQ, HID] bf16
    const bf16_t* __restrict__ pwb,    // [HID, HID] bf16
    const float*  __restrict__ bias,   // [HID]
    const float*  __restrict__ resid,  // [SEQ, HID]
    float* __restrict__ out)           // [SEQ, HID]
{
    const int lane = threadIdx.x & 31;
    const int w    = threadIdx.x >> 5;
    const int l16  = lane & 15, lhi = lane >> 4;
    const int mb   = (blockIdx.y * 2 + (w >> 2)) * 16;
    const int nb   = (blockIdx.x * 4 + (w & 3)) * 16;

    const bf16_t* arow = ab  + (size_t)(mb + l16) * HID;
    const bf16_t* brow = pwb + (size_t)(nb + l16) * HID;

    v8f acc = {};
#pragma unroll 4
    for (int kb = 0; kb < HID; kb += KSTEP) {
        __builtin_prefetch(brow + kb + 8 * KSTEP, 0, 0);
        v8bf alo = *(const v8bf*)(arow + kb + lhi * 8);
        v8bf ahi = *(const v8bf*)(arow + kb + 16 + lhi * 8);
        v8bf blo = *(const v8bf*)(brow + kb + lhi * 8);
        v8bf bhi = *(const v8bf*)(brow + kb + 16 + lhi * 8);
        acc = WMMA_BF16(combine16(alo, ahi), combine16(blo, bhi), acc);
    }

    const int n = nb + l16;
    const float bn = bias[n];
#pragma unroll
    for (int r = 0; r < 8; ++r) {
        const int s = mb + r + lhi * 8;
        out[(size_t)s * HID + n] = acc[r] + bn + resid[(size_t)s * HID + n];
    }
}

// ---------------------------------------------------------------------------
// Host launcher
// ---------------------------------------------------------------------------
extern "C" void kernel_launch(void* const* d_in, const int* in_sizes, int n_in,
                              void* d_out, int out_size, void* d_ws, size_t ws_size,
                              hipStream_t stream) {
    const float* hidden = (const float*)d_in[0];  // [S, HID]
    const float* xnorm  = (const float*)d_in[1];  // [S, HID]
    const float* qkv_w  = (const float*)d_in[2];  // [3*HID, HID]
    const float* qkv_b  = (const float*)d_in[3];  // [3*HID]
    const float* proj_w = (const float*)d_in[4];  // [HID, HID]
    const float* proj_b = (const float*)d_in[5];  // [HID]
    const float* cosT   = (const float*)d_in[6];  // [S, HD]
    const float* sinT   = (const float*)d_in[7];  // [S, HD]
    float* out = (float*)d_out;

    // workspace carve-up (256B aligned)
    char* base = (char*)d_ws;
    size_t off = 0;
    auto carve = [&](size_t bytes) {
        char* p = base + off;
        off += (bytes + 255) & ~(size_t)255;
        return p;
    };
    bf16_t* xb   = (bf16_t*)carve((size_t)SEQ * HID * 2);
    bf16_t* wb   = (bf16_t*)carve((size_t)3 * HID * HID * 2);
    bf16_t* pwb  = (bf16_t*)carve((size_t)HID * HID * 2);
    bf16_t* qp   = (bf16_t*)carve((size_t)NH * SEQ * DPAD * 2);
    bf16_t* kpb  = (bf16_t*)carve((size_t)NH * SEQ * DPAD * 2);
    bf16_t* vt   = (bf16_t*)carve((size_t)NH * HD * SEQ * 2);
    bf16_t* attn = (bf16_t*)carve((size_t)SEQ * HID * 2);
    (void)ws_size; (void)in_sizes; (void)n_in; (void)out_size;

    // 1) convert operands to bf16
    {
        int n;
        n = SEQ * HID;
        cvt_f32_bf16<<<(n + 255) / 256, 256, 0, stream>>>(xnorm, xb, n);
        n = 3 * HID * HID;
        cvt_f32_bf16<<<(n + 255) / 256, 256, 0, stream>>>(qkv_w, wb, n);
        n = HID * HID;
        cvt_f32_bf16<<<(n + 255) / 256, 256, 0, stream>>>(proj_w, pwb, n);
    }

    // 2) QKV GEMM  (N tiles: 3840/64 = 60, M tiles: 3136/32 = 98)
    qkv_gemm<<<dim3(60, 98), 256, 0, stream>>>(xb, wb, qkv_b, qp, kpb, vt);

    // 3) RoPE + pad-zero  (2 * NH * SEQ * 48 threads)
    {
        size_t total = (size_t)2 * NH * SEQ * 48;
        rope_kernel<<<(unsigned)((total + 255) / 256), 256, 0, stream>>>(qp, kpb, cosT, sinT);
    }

    // 4) flash attention  (49 row-blocks x 16 heads, 4 waves each)
    attn_kernel<<<dim3(49, NH), 128, 0, stream>>>(qp, kpb, vt, attn);

    // 5) projection + bias + residual  (N tiles: 1280/64 = 20, M: 98)
    proj_gemm<<<dim3(20, 98), 256, 0, stream>>>(attn, pwb, proj_b, hidden, out);
}
